// MM_LSTM_Age_v2_old_70179765616735
// MI455X (gfx1250) — compile-verified, matrix operand order checked by reading, not run on hardware
//
#include <hip/hip_runtime.h>
#include <hip/hip_bf16.h>

// ---------------------------------------------------------------------------
// MI455X (gfx1250) implementation.
// 22 LSTMs (H=256, B=64, T=365). All GEMMs use v_wmma_f32_16x16x32_bf16.
// Weights are packed once per call into WMMA B-fragment layout (bf16) so each
// lane's fragment is a contiguous 32B load. Recurrent state lives in LDS
// (h double-buffered bf16) + registers (c). x_t tiles are staged into LDS via
// GLOBAL_LOAD_ASYNC_TO_LDS_B128 (ASYNCcnt) each step. One workgroup
// (32 waves, wave32) per LSTM instance; phases: eco(1 WG) -> data/att(14 WG)
// -> prod mul -> ts(7 WG) -> MLP head.
// ---------------------------------------------------------------------------

typedef __attribute__((ext_vector_type(16))) __bf16 v16bf;
typedef __attribute__((ext_vector_type(8)))  float  v8f;

#define Bsz   64
#define Tlen  365
#define Hdim  256
#define G4    1024
#define NOUT  7
#define DIN   42      // N_CONS(20) + N_OUT(7) + N_AGE(15)

__device__ __forceinline__ float sigm_(float x) { return 1.0f / (1.0f + __expf(-x)); }

__device__ __forceinline__ void wait_async0() {
#if __has_builtin(__builtin_amdgcn_s_wait_asynccnt)
  __builtin_amdgcn_s_wait_asynccnt(0);
#else
  asm volatile("s_wait_asynccnt 0x0" ::: "memory");
#endif
}

// --------------------------- weight packing --------------------------------
// Packed layout: Wp[kt][nt][lane][16 halves], 32B contiguous per lane.
// lane 0-15 : col n = nt*16 + lane,      K = kt*32 + {0..7, 16..23}
// lane 16-31: col n = nt*16 + (lane-16), K = kt*32 + {8..15, 24..31}
// K rows [0, dpad) come from Wk (zero-padded past dk), rows [dpad, dpad+256)
// come from Wr. Both sources are f32 row-major [K][1024].
__global__ void pack_w(const float* __restrict__ Wk, int dk, int dpad,
                       const float* __restrict__ Wr,
                       __bf16* __restrict__ Wp, int KT) {
  int n = KT * 64 * 32 * 16;
  for (int idx = blockIdx.x * blockDim.x + threadIdx.x; idx < n;
       idx += gridDim.x * blockDim.x) {
    int hh   = idx & 15;
    int lane = (idx >> 4) & 31;
    int nt   = (idx >> 9) & 63;
    int kt   = idx >> 15;
    int col  = nt * 16 + (lane & 15);
    int kb   = kt * 32 + ((lane & 16) ? 8 : 0);
    int k    = kb + (hh < 8 ? hh : hh + 8);
    float v;
    if (k < dpad) v = (k < dk) ? Wk[(size_t)k * G4 + col] : 0.0f;
    else          v = Wr[(size_t)(k - dpad) * G4 + col];
    Wp[idx] = (__bf16)v;
  }
}

// --------------------------- input staging ---------------------------------
// A_eco: [B*T][64] bf16 = concat(x_eco(20), x_age(15)), zero pad to 64.
__global__ void prep_eco(const float* __restrict__ in, __bf16* __restrict__ A) {
  int n = Bsz * Tlen * 64;
  for (int idx = blockIdx.x * blockDim.x + threadIdx.x; idx < n;
       idx += gridDim.x * blockDim.x) {
    int k  = idx & 63;
    int bt = idx >> 6;
    float v = 0.0f;
    if (k < 20)      v = in[(size_t)bt * DIN + k];
    else if (k < 35) v = in[(size_t)bt * DIN + 27 + (k - 20)];
    A[idx] = (__bf16)v;
  }
}

// curin: [7][B][T][288] bf16. Static cols 0..3 + zero pad 260..287.
// Cols 4..259 (x_ecos) are written later by the eco recurrent kernel.
__global__ void prep_curin(const float* __restrict__ in, __bf16* __restrict__ cu) {
  int n = NOUT * Bsz * Tlen;
  for (int idx = blockIdx.x * blockDim.x + threadIdx.x; idx < n;
       idx += gridDim.x * blockDim.x) {
    int t  = idx % Tlen;
    int ib = idx / Tlen;
    int b  = ib % Bsz;
    int i  = ib / Bsz;
    const float* s = in + ((size_t)b * Tlen + t) * DIN;
    __bf16* d = cu + (size_t)idx * 288;
    d[0] = (__bf16)s[20 + i];        // init_i
    d[1] = (__bf16)s[27 + 0];        // age col 0
    d[2] = (__bf16)s[27 + i + 1];    // age col i+1
    d[3] = (__bf16)s[27 + i + 8];    // age col i+8
    for (int k = 260; k < 288; ++k) d[k] = (__bf16)0.0f;
  }
}

__global__ void mul_hs(const __bf16* __restrict__ a, const __bf16* __restrict__ b,
                       __bf16* __restrict__ p, int n) {
  for (int idx = blockIdx.x * blockDim.x + threadIdx.x; idx < n;
       idx += gridDim.x * blockDim.x)
    p[idx] = (__bf16)((float)a[idx] * (float)b[idx]);
}

// --------------------------- fragment loads --------------------------------
// A fragment (16x32 bf16) from LDS: rows striped per-lane (ISA 7.12.2).
__device__ __forceinline__ v16bf frag_a(const __bf16* __restrict__ sX, int xdim,
                                        const __bf16* __restrict__ sHr,
                                        int row0, int k0, int lane) {
  int r  = row0 + (lane & 15);
  int kb = (lane & 16) ? 8 : 0;
  const __bf16* p;
  if (k0 < xdim) p = sX + (size_t)r * xdim + (k0 + kb);         // x region
  else           p = sHr + (size_t)r * Hdim + (k0 - xdim + kb); // h region
  v16bf f;
#pragma unroll
  for (int i = 0; i < 8; ++i) f[i] = p[i];
#pragma unroll
  for (int i = 0; i < 8; ++i) f[8 + i] = p[16 + i];
  return f;
}

// --------------------------- recurrent kernel ------------------------------
// mode 0: eco    (1 WG)  -> scatter h into all 7 curin buffers (cols 4..259)
// mode 1: data/att (14 WG: 0..6 data tanh, 7..13 att sigmoid) -> full h seq
// mode 2: ts     (7 WG)  -> final h (f32)
// gates = [x_t , h_{t-1}] @ Wp + bias ; K = xdim + 256, all multiples of 32.
__global__ __launch_bounds__(1024, 1)
void lstm_rec(const __bf16* __restrict__ xin, int xdim, int KT,
              const __bf16* __restrict__ Wp, size_t WpStrideEl,
              const float* __restrict__ bias0,
              const float* __restrict__ bias1,
              int mode,
              __bf16* __restrict__ outCur,   // mode 0
              __bf16* __restrict__ hs0,      // mode 1 (data)
              __bf16* __restrict__ hs1,      // mode 1 (att)
              float*  __restrict__ outF) {   // mode 2
  const int bi   = blockIdx.x;
  const int lane = threadIdx.x & 31;
  const int wave = threadIdx.x >> 5;

  int actg = 0;
  const float*  bias = bias0;
  const __bf16* x    = xin;
  const __bf16* W    = Wp;
  __bf16* hs = nullptr;
  float*  fh = nullptr;

  if (mode == 1) {
    int branch = bi % NOUT;
    int isAtt  = bi / NOUT;
    actg = isAtt;
    bias = (isAtt ? bias1 : bias0) + branch * G4;
    x    = xin + (size_t)branch * Bsz * Tlen * xdim;
    W    = Wp + (size_t)bi * WpStrideEl;
    hs   = (isAtt ? hs1 : hs0) + (size_t)branch * Bsz * Tlen * Hdim;
  } else if (mode == 2) {
    bias = bias0 + bi * G4;
    x    = xin + (size_t)bi * Bsz * Tlen * xdim;
    W    = Wp + (size_t)bi * WpStrideEl;
    fh   = outF + (size_t)bi * Bsz * Hdim;
  }

  extern __shared__ __align__(16) char smem[];
  __bf16* sX = (__bf16*)smem;                 // [64][xdim]
  __bf16* sH = sX + (size_t)Bsz * xdim;       // [2][64][256] double buffer
  const int HS = Bsz * Hdim;                  // 16384

  for (int i = threadIdx.x; i < HS; i += blockDim.x) sH[i] = (__bf16)0.0f;

  float creg[2][8];                           // cell state, owned per lane
#pragma unroll
  for (int it = 0; it < 2; ++it)
#pragma unroll
    for (int r = 0; r < 8; ++r) creg[it][r] = 0.0f;

  const int nrow8 = xdim >> 3;                // 16B chunks per row
  const int n8    = Bsz * nrow8;

  for (int t = 0; t < Tlen; ++t) {
    // stage x_t into LDS via async DMA (no VGPR round trip, ASYNCcnt);
    // prefetch x_{t+1} toward L2/WGP$
    for (int i = threadIdx.x; i < n8; i += blockDim.x) {
      int b = i / nrow8, c = i % nrow8;
      const __bf16* src = x + ((size_t)b * Tlen + t) * xdim + (size_t)c * 8;
      unsigned lds = (unsigned)(uintptr_t)((char*)sX + (size_t)i * 16);
      unsigned long long ga = (unsigned long long)(uintptr_t)src;
      asm volatile("global_load_async_to_lds_b128 %0, %1, off"
                   :: "v"(lds), "v"(ga) : "memory");
      if (t + 1 < Tlen)
        __builtin_prefetch((const void*)(src + xdim), 0, 1);
    }
    wait_async0();
    __syncthreads();

    const __bf16* sHr = sH + (t & 1) * HS;
    __bf16*       sHw = sH + ((t + 1) & 1) * HS;

#pragma unroll
    for (int it = 0; it < 2; ++it) {          // 64 tile-pairs over 32 waves
      int pair = wave + it * 32;
      int m  = pair & 3;                      // row tile (B/16)
      int jt = pair >> 2;                     // hidden-col tile (H/16)

      // 4 gate accumulators (i,f,g,o), init from bias
      v8f acc[4];
#pragma unroll
      for (int g = 0; g < 4; ++g) {
        float bw = bias[g * Hdim + jt * 16 + (lane & 15)];
#pragma unroll
        for (int r = 0; r < 8; ++r) acc[g][r] = bw;
      }

      for (int kt = 0; kt < KT; ++kt) {
        v16bf afr = frag_a(sX, xdim, sHr, m * 16, kt * 32, lane);
        const v16bf* wb = (const v16bf*)W + ((size_t)kt * 64) * 32;
#pragma unroll
        for (int g = 0; g < 4; ++g) {
          int nt = g * 16 + jt;
          v16bf bfr = wb[(size_t)nt * 32 + lane];
          acc[g] = __builtin_amdgcn_wmma_f32_16x16x32_bf16(
              false, afr, false, bfr, (short)0, acc[g], false, false);
        }
      }

      // activations + state update (C/D layout: VGPR r -> row m*16+r(+8))
      int j      = jt * 16 + (lane & 15);
      int rowAdd = (lane & 16) ? 8 : 0;
#pragma unroll
      for (int r = 0; r < 8; ++r) {
        int row = m * 16 + r + rowAdd;
        float zi = acc[0][r], zf = acc[1][r], zg = acc[2][r], zo = acc[3][r];
        float gv = actg ? sigm_(zg) : tanhf(zg);
        float cn = sigm_(zf) * creg[it][r] + sigm_(zi) * gv;
        float hv = sigm_(zo) * (actg ? sigm_(cn) : tanhf(cn));
        creg[it][r] = cn;
        sHw[row * Hdim + j] = (__bf16)hv;
        if (mode == 0) {
#pragma unroll
          for (int br = 0; br < NOUT; ++br)
            outCur[(((size_t)(br * Bsz + row)) * Tlen + t) * 288 + 4 + j] =
                (__bf16)hv;
        } else if (mode == 1) {
          hs[((size_t)row * Tlen + t) * Hdim + j] = (__bf16)hv;
        } else if (t == Tlen - 1) {
          fh[row * Hdim + j] = hv;
        }
      }
    }
    __syncthreads();
  }
}

// --------------------------- MLP head --------------------------------------
__global__ void head_kernel(const float* __restrict__ hT,
                            const float* __restrict__ d1W, const float* __restrict__ d1b,
                            const float* __restrict__ d2W, const float* __restrict__ d2b,
                            const float* __restrict__ dfW, const float* __restrict__ dfb,
                            const float* __restrict__ in, float* __restrict__ out) {
  int i = blockIdx.x, tid = threadIdx.x;
  __shared__ float sh[256], s1[256], s2[64];
  const float* W1 = d1W + (size_t)i * 256 * 256;
  const float* W2 = d2W + (size_t)i * 256 * 64;
  const float* Wf = dfW + (size_t)i * 67;
  for (int b = 0; b < Bsz; ++b) {
    sh[tid] = hT[((size_t)i * Bsz + b) * Hdim + tid];
    __syncthreads();
    float a = d1b[i * 256 + tid];
    for (int j = 0; j < 256; ++j) a += sh[j] * W1[j * 256 + tid];
    s1[tid] = fmaxf(a, 0.0f);
    __syncthreads();
    if (tid < 64) {
      float a2 = d2b[i * 64 + tid];
      for (int j = 0; j < 256; ++j) a2 += s1[j] * W2[j * 64 + tid];
      s2[tid] = fmaxf(a2, 0.0f);
    }
    __syncthreads();
    if (tid == 0) {
      float o = dfb[i];
      for (int j = 0; j < 64; ++j) o += s2[j] * Wf[j];
      const float* last = in + ((size_t)b * Tlen + (Tlen - 1)) * DIN;
      o += last[27 + 0]     * Wf[64];
      o += last[27 + i + 1] * Wf[65];
      o += last[27 + i + 8] * Wf[66];
      o += last[20 + i];                         // residual init_i
      out[b * NOUT + i] = o;
    }
    __syncthreads();
  }
}

// --------------------------- host driver -----------------------------------
extern "C" void kernel_launch(void* const* d_in, const int* in_sizes, int n_in,
                              void* d_out, int out_size, void* d_ws, size_t ws_size,
                              hipStream_t stream) {
  (void)in_sizes; (void)n_in; (void)out_size; (void)ws_size;
  const float* inputs = (const float*)d_in[0];
  const float* ecoWk  = (const float*)d_in[1];
  const float* ecoWr  = (const float*)d_in[2];
  const float* ecoB   = (const float*)d_in[3];
  const float* dWk    = (const float*)d_in[4];
  const float* dWr    = (const float*)d_in[5];
  const float* dB     = (const float*)d_in[6];
  const float* aWk    = (const float*)d_in[7];
  const float* aWr    = (const float*)d_in[8];
  const float* aB     = (const float*)d_in[9];
  const float* tWk    = (const float*)d_in[10];
  const float* tWr    = (const float*)d_in[11];
  const float* tB     = (const float*)d_in[12];
  const float* d1W    = (const float*)d_in[13];
  const float* d1b    = (const float*)d_in[14];
  const float* d2W    = (const float*)d_in[15];
  const float* d2b    = (const float*)d_in[16];
  const float* dfW    = (const float*)d_in[17];
  const float* dfb    = (const float*)d_in[18];

  // workspace carve-up (256B aligned)
  char* w = (char*)d_ws;
  size_t off = 0;
  auto take = [&](size_t bytes) -> char* {
    char* p = w + off;
    off = (off + bytes + 255) & ~(size_t)255;
    return p;
  };
  const size_t TILE = 32768;                       // elements per kt (64*32*16)
  __bf16* WpEco = (__bf16*)take(10 * TILE * 2);
  __bf16* WpDA  = (__bf16*)take((size_t)14 * 17 * TILE * 2);
  __bf16* WpTs  = (__bf16*)take((size_t)7 * 16 * TILE * 2);
  __bf16* Aeco  = (__bf16*)take((size_t)Bsz * Tlen * 64 * 2);
  __bf16* curin = (__bf16*)take((size_t)NOUT * Bsz * Tlen * 288 * 2);
  size_t hsEl   = (size_t)NOUT * Bsz * Tlen * Hdim;
  __bf16* dhs   = (__bf16*)take(hsEl * 2);
  __bf16* ahs   = (__bf16*)take(hsEl * 2);
  __bf16* prod  = (__bf16*)take(hsEl * 2);
  float*  hT    = (float*)take((size_t)NOUT * Bsz * Hdim * 4);

  // 1) pack weights into WMMA fragment layout (bf16)
  pack_w<<<256, 256, 0, stream>>>(ecoWk, 35, 64, ecoWr, WpEco, 10);
  for (int i = 0; i < NOUT; ++i) {
    pack_w<<<256, 256, 0, stream>>>(dWk + (size_t)i * 260 * G4, 260, 288,
                                    dWr + (size_t)i * 256 * G4,
                                    WpDA + (size_t)i * 17 * TILE, 17);
    pack_w<<<256, 256, 0, stream>>>(aWk + (size_t)i * 260 * G4, 260, 288,
                                    aWr + (size_t)i * 256 * G4,
                                    WpDA + (size_t)(NOUT + i) * 17 * TILE, 17);
    pack_w<<<256, 256, 0, stream>>>(tWk + (size_t)i * 256 * G4, 256, 256,
                                    tWr + (size_t)i * 256 * G4,
                                    WpTs + (size_t)i * 16 * TILE, 16);
  }

  // 2) stage inputs
  prep_eco<<<1024, 256, 0, stream>>>(inputs, Aeco);
  prep_curin<<<512, 256, 0, stream>>>(inputs, curin);

  // 3) eco LSTM (scatter x_ecos into all 7 curin buffers)
  size_t shEco = ((size_t)Bsz * 64 + 2 * Bsz * Hdim) * 2;       // 73,728 B
  lstm_rec<<<1, 1024, shEco, stream>>>(Aeco, 64, 10, WpEco, 0, ecoB, nullptr,
                                       0, curin, nullptr, nullptr, nullptr);

  // 4) data + att LSTMs, 14-wide
  size_t shBr = ((size_t)Bsz * 288 + 2 * Bsz * Hdim) * 2;       // 102,400 B
  lstm_rec<<<14, 1024, shBr, stream>>>(curin, 288, 17, WpDA, (size_t)17 * TILE,
                                       dB, aB, 1, nullptr, dhs, ahs, nullptr);

  // 5) prod = data_hs * att_hs
  mul_hs<<<4096, 256, 0, stream>>>(dhs, ahs, prod, (int)hsEl);

  // 6) ts LSTMs, 7-wide (final h only)
  size_t shTs = ((size_t)Bsz * 256 + 2 * Bsz * Hdim) * 2;       // 98,304 B
  lstm_rec<<<7, 1024, shTs, stream>>>(prod, 256, 16, WpTs, (size_t)16 * TILE,
                                      tB, nullptr, 2, nullptr, nullptr, nullptr, hT);

  // 7) MLP head + residual -> out [B, NOUT] f32
  head_kernel<<<NOUT, 256, 0, stream>>>(hT, d1W, d1b, d2W, d2b, dfW, dfb,
                                        inputs, (float*)d_out);
}
